// BaseModel_62122406969971
// MI455X (gfx1250) — matrix operand order repaired and checked by reading, not compiled, not run on hardware
//
#include <hip/hip_runtime.h>
#include <hip/hip_bf16.h>

#define HDIM 128
#define EDGES_PER_BLOCK 128
#define MAX_BLOCKS 2048

// padded LDS strides (elements): +8 bf16 -> row shift of 4 banks, conflict-free
#define W1K 264   // 256 + 8
#define W2K 136   // 128 + 8

typedef __attribute__((ext_vector_type(16))) __bf16 v16bf;
typedef __attribute__((ext_vector_type(8)))  __bf16 v8bf;
typedef __attribute__((ext_vector_type(2)))  __bf16 v2bf;
typedef __attribute__((ext_vector_type(8)))  float  v8f;

__device__ __forceinline__ __bf16 f2bf(float f) { return (__bf16)f; }

// pack two f32 into one dword of bf16 pair (lowers to v_cvt_pk_bf16_f32)
__device__ __forceinline__ unsigned pk2bf(float a, float b) {
  union { v2bf v; unsigned u; } t;
  t.v[0] = (__bf16)a; t.v[1] = (__bf16)b;
  return t.u;
}

__device__ __forceinline__ v8f wmma_bf16(v16bf a, v16bf b, v8f c) {
  return __builtin_amdgcn_wmma_f32_16x16x32_bf16(false, a, false, b, (short)0, c,
                                                 false, false);
}

__global__ __launch_bounds__(256, 4) void edge_mlp_kernel(
    const float* __restrict__ z,
    const long long* __restrict__ edge,
    const float* __restrict__ W1, const float* __restrict__ b1,
    const float* __restrict__ W2, const float* __restrict__ b2,
    const float* __restrict__ W3, const float* __restrict__ b3,
    float* __restrict__ out, int nTiles, int E)
{
  // transposed GEMMs: H1^T = W1^T x emb^T, H2^T = W2^T x H1^T
  // -> only weights live in LDS; activations never leave registers.
  __shared__ __align__(16) __bf16 sW1[128 * W1K];  // W1^T [hidden][k] bf16
  __shared__ __align__(16) __bf16 sW2[64  * W2K];  // W2^T [hidden][k] bf16
  __shared__ __align__(16) float  sB1[128];
  __shared__ __align__(16) float  sB2[64];
  __shared__ __align__(16) float  sW3v[64];

  const int tid = threadIdx.x;

  // ---- stage weights: transpose + convert to bf16 (coalesced global reads) ----
  for (int i = tid; i < 256 * 128; i += 256) {
    const int k = i >> 7, n = i & 127;            // W1 is (256,128) row-major
    sW1[n * W1K + k] = f2bf(W1[i]);
  }
  for (int i = tid; i < 128 * 64; i += 256) {
    const int k = i >> 6, n = i & 63;             // W2 is (128,64) row-major
    sW2[n * W2K + k] = f2bf(W2[i]);
  }
  if (tid < 128) sB1[tid] = b1[tid];
  if (tid < 64)  sB2[tid] = b2[tid];
  if (tid < 64)  sW3v[tid] = W3[tid];
  __syncthreads();

  const float b3v = b3[0];
  const int wave = tid >> 5;
  const int lane = tid & 31;
  const int h    = lane >> 4;     // lane group 0/1
  const int lrow = lane & 15;
  const int klow = h << 3;        // A-layout interleave: hi group starts at K+8
  const int n0   = wave << 4;     // this wave's 16-edge slice

  for (int tile = blockIdx.x; tile < nTiles; tile += gridDim.x) {
    // compiler barrier: keep LDS weight-fragment loads INSIDE the loop.
    // (without it, LICM hoists 640+ VGPRs of fragments and spills to scratch)
    asm volatile("" ::: "memory");

    const int rowIdx = tile * EDGES_PER_BLOCK + n0 + lrow;  // this lane's edge
    const int eClamp = rowIdx < E ? rowIdx : (E - 1);
    const long long s = edge[2 * (long long)eClamp];
    const long long d = edge[2 * (long long)eClamp + 1];
    const float* __restrict__ srcRow = z + (size_t)s * HDIM;
    const float* __restrict__ dstRow = z + (size_t)d * HDIM;

    // ============ layer 1: H1^T(128x16) = W1^T(128x256) x emb^T(256x16) ======
    v8f C1[8] = {};   // 8 hidden M-tiles; D: col = edge(lane), row = hidden
    #pragma unroll
    for (int kc = 0; kc < 8; ++kc) {
      // B fragment (edge data): lane = its edge, k = h*8+0..7 and 16+h*8+0..7
      const float* rp = (kc < 4) ? srcRow : dstRow;
      const int cb = ((kc & 3) << 5) + (h << 3);
      const float4 f0 = *(const float4*)(rp + cb);
      const float4 f1 = *(const float4*)(rp + cb + 4);
      const float4 f2 = *(const float4*)(rp + cb + 16);
      const float4 f3 = *(const float4*)(rp + cb + 20);
      v16bf b;
      b[0]=f2bf(f0.x);  b[1]=f2bf(f0.y);  b[2]=f2bf(f0.z);  b[3]=f2bf(f0.w);
      b[4]=f2bf(f1.x);  b[5]=f2bf(f1.y);  b[6]=f2bf(f1.z);  b[7]=f2bf(f1.w);
      b[8]=f2bf(f2.x);  b[9]=f2bf(f2.y);  b[10]=f2bf(f2.z); b[11]=f2bf(f2.w);
      b[12]=f2bf(f3.x); b[13]=f2bf(f3.y); b[14]=f2bf(f3.z); b[15]=f2bf(f3.w);

      #pragma unroll
      for (int j = 0; j < 8; ++j) {
        // A fragment (weights, interleaved layout) from LDS
        const __bf16* ap = &sW1[(j * 16 + lrow) * W1K + (kc << 5) + klow];
        const v8bf alo = *(const v8bf*)ap;
        const v8bf ahi = *(const v8bf*)(ap + 16);
        const v16bf a = __builtin_shufflevector(alo, ahi,
            0,1,2,3,4,5,6,7,8,9,10,11,12,13,14,15);
        C1[j] = wmma_bf16(a, b, C1[j]);
      }
    }

    // ---- epilogue 1: bias + ReLU + pack to bf16 dwords (stay in registers) ----
    // lane holds hidden m = j*16 + 8*h + v of its edge
    unsigned P[8][4];
    #pragma unroll
    for (int j = 0; j < 8; ++j) {
      const float4 ba = *(const float4*)&sB1[j * 16 + h * 8];
      const float4 bb = *(const float4*)&sB1[j * 16 + h * 8 + 4];
      float x0 = C1[j][0] + ba.x; x0 = x0 > 0.f ? x0 : 0.f;
      float x1 = C1[j][1] + ba.y; x1 = x1 > 0.f ? x1 : 0.f;
      float x2 = C1[j][2] + ba.z; x2 = x2 > 0.f ? x2 : 0.f;
      float x3 = C1[j][3] + ba.w; x3 = x3 > 0.f ? x3 : 0.f;
      float x4 = C1[j][4] + bb.x; x4 = x4 > 0.f ? x4 : 0.f;
      float x5 = C1[j][5] + bb.y; x5 = x5 > 0.f ? x5 : 0.f;
      float x6 = C1[j][6] + bb.z; x6 = x6 > 0.f ? x6 : 0.f;
      float x7 = C1[j][7] + bb.w; x7 = x7 > 0.f ? x7 : 0.f;
      P[j][0] = pk2bf(x0, x1); P[j][1] = pk2bf(x2, x3);
      P[j][2] = pk2bf(x4, x5); P[j][3] = pk2bf(x6, x7);
    }

    // ============ layer 2: H2^T(64x16) = W2^T(64x128) x H1^T(128x16) =========
    // B fragment assembled in registers: lane needs k = kc2*32 + h*16 + 0..15 of
    // its own edge; half comes from itself, half from partner lane (lane^16).
    v8f C2[4] = {};
    #pragma unroll
    for (int kc2 = 0; kc2 < 4; ++kc2) {
      const int tA = 2 * kc2, tB = tA + 1;
      unsigned Xa[4], Xb[4];
      #pragma unroll
      for (int q = 0; q < 4; ++q) {
        Xa[q] = (unsigned)__shfl_xor((int)P[tA][q], 16);
        Xb[q] = (unsigned)__shfl_xor((int)P[tB][q], 16);
      }
      union { unsigned u[8]; v16bf v; } B2;
      #pragma unroll
      for (int q = 0; q < 4; ++q) {
        B2.u[q]     = h ? Xb[q]     : P[tA][q];   // k = 16*t0 + 0..7
        B2.u[4 + q] = h ? P[tB][q]  : Xa[q];      // k = 16*t0 + 8..15
      }
      #pragma unroll
      for (int mt = 0; mt < 4; ++mt) {
        const __bf16* ap = &sW2[(mt * 16 + lrow) * W2K + (kc2 << 5) + (h << 4)];
        const v8bf alo = *(const v8bf*)ap;
        const v8bf ahi = *(const v8bf*)(ap + 8);
        const v16bf a = __builtin_shufflevector(alo, ahi,
            0,1,2,3,4,5,6,7,8,9,10,11,12,13,14,15);
        C2[mt] = wmma_bf16(a, B2.v, C2[mt]);
      }
    }

    // ---- epilogue 2 + layer 3: bias + ReLU + dot(W3), all in registers ----
    // lane holds h2 values m = mt*16 + 8*h + v of its edge
    float acc = 0.f;
    #pragma unroll
    for (int mt = 0; mt < 4; ++mt) {
      const float4 ba = *(const float4*)&sB2[mt * 16 + h * 8];
      const float4 bb = *(const float4*)&sB2[mt * 16 + h * 8 + 4];
      const float4 wa = *(const float4*)&sW3v[mt * 16 + h * 8];
      const float4 wb = *(const float4*)&sW3v[mt * 16 + h * 8 + 4];
      float t;
      t = C2[mt][0] + ba.x; t = t > 0.f ? t : 0.f; acc = fmaf(t, wa.x, acc);
      t = C2[mt][1] + ba.y; t = t > 0.f ? t : 0.f; acc = fmaf(t, wa.y, acc);
      t = C2[mt][2] + ba.z; t = t > 0.f ? t : 0.f; acc = fmaf(t, wa.z, acc);
      t = C2[mt][3] + ba.w; t = t > 0.f ? t : 0.f; acc = fmaf(t, wa.w, acc);
      t = C2[mt][4] + bb.x; t = t > 0.f ? t : 0.f; acc = fmaf(t, wb.x, acc);
      t = C2[mt][5] + bb.y; t = t > 0.f ? t : 0.f; acc = fmaf(t, wb.y, acc);
      t = C2[mt][6] + bb.z; t = t > 0.f ? t : 0.f; acc = fmaf(t, wb.z, acc);
      t = C2[mt][7] + bb.w; t = t > 0.f ? t : 0.f; acc = fmaf(t, wb.w, acc);
    }
    // partner lane holds the complementary hidden half of the same edge
    acc += __shfl_xor(acc, 16);
    if (h == 0 && rowIdx < E) out[rowIdx] = acc + b3v;  // coalesced 16-lane store
  }
}

extern "C" void kernel_launch(void* const* d_in, const int* in_sizes, int n_in,
                              void* d_out, int out_size, void* d_ws, size_t ws_size,
                              hipStream_t stream) {
  const float*     z    = (const float*)d_in[0];
  const long long* edge = (const long long*)d_in[1];   // int64 per reference
  const float*     W1   = (const float*)d_in[2];
  const float*     b1   = (const float*)d_in[3];
  const float*     W2   = (const float*)d_in[4];
  const float*     b2   = (const float*)d_in[5];
  const float*     W3   = (const float*)d_in[6];
  const float*     b3   = (const float*)d_in[7];
  float* out = (float*)d_out;

  const int E = out_size;                                // 1,000,000 edges
  const int nTiles = (E + EDGES_PER_BLOCK - 1) / EDGES_PER_BLOCK;
  const int grid = nTiles < MAX_BLOCKS ? nTiles : MAX_BLOCKS;
  edge_mlp_kernel<<<grid, 256, 0, stream>>>(z, edge, W1, b1, W2, b2, W3, b3,
                                            out, nTiles, E);
}
